// Pseudo_Loss_17368847745317
// MI455X (gfx1250) — matrix operand order repaired and checked by reading, not compiled, not run on hardware
//
#include <hip/hip_runtime.h>

// K-means (100 iters) + pseudo-label cross-entropy for MI455X (gfx1250).
// GEMM core: v_wmma_f32_16x16x32_bf16, c staged in LDS, x resident in L2.
// Segment-sum pre-accumulated in 128KB LDS (ds_add_f32) -- uses CDNA5's
// 320KB/WGP LDS to cut global atomic traffic 4x.

#define KCL   128
#define DIM   256
#define NPT   32768
#define ITERS 100

typedef __attribute__((ext_vector_type(16))) __bf16 v16bf;
typedef __attribute__((ext_vector_type(8)))  float  v8f;

union Frag {
    v16bf v;
    uint4 q[2];
    unsigned int u[8];
};

__device__ __forceinline__ unsigned int pack2bf(float a, float b) {
    // round-to-nearest-even f32 -> bf16, packed lo|hi
    unsigned int ua = __builtin_bit_cast(unsigned int, a);
    unsigned int ub = __builtin_bit_cast(unsigned int, b);
    ua = ua + 0x7FFFu + ((ua >> 16) & 1u);
    ub = ub + 0x7FFFu + ((ub >> 16) & 1u);
    return (ua >> 16) | (ub & 0xFFFF0000u);
}

__device__ __forceinline__ unsigned short f2bf(float a) {
    unsigned int u = __builtin_bit_cast(unsigned int, a);
    u = u + 0x7FFFu + ((u >> 16) & 1u);
    return (unsigned short)(u >> 16);
}

__device__ __forceinline__ float bf2f(unsigned short h) {
    unsigned int u = ((unsigned int)h) << 16;
    return __builtin_bit_cast(float, u);
}

// Load one 16x32 bf16 A-fragment (contiguous k order, same as B) from f32 row, converting on the fly.
// Lane half h: element e -> K = h*16 + e.
__device__ __forceinline__ void load_a_frag(Frag& f, const float* xr, int h) {
    const float4* p = (const float4*)(xr + h * 16);
    const float4* q = (const float4*)(xr + h * 16 + 8);
    float4 a0 = p[0], a1 = p[1], a2 = q[0], a3 = q[1];
    f.u[0] = pack2bf(a0.x, a0.y); f.u[1] = pack2bf(a0.z, a0.w);
    f.u[2] = pack2bf(a1.x, a1.y); f.u[3] = pack2bf(a1.z, a1.w);
    f.u[4] = pack2bf(a2.x, a2.y); f.u[5] = pack2bf(a2.z, a2.w);
    f.u[6] = pack2bf(a3.x, a3.y); f.u[7] = pack2bf(a3.z, a3.w);
}

// ---------------------------------------------------------------------------
__global__ void k_init(float* sums, float* counts, float* out) {
    int i = blockIdx.x * 256 + threadIdx.x;
    if (i < KCL * DIM) sums[i] = 0.0f;
    if (i < KCL)       counts[i] = 0.0f;
    if (i == 0)        out[0] = 0.0f;
}

// c0 = x[init_idx]; emit bf16 centroids + 0.5*||c||^2 (computed on bf16 values).
__global__ void k_gather(const float* __restrict__ x, const int* __restrict__ idx,
                         unsigned short* __restrict__ cb, float* __restrict__ hn) {
    __shared__ float red[256];
    int k = blockIdx.x, d = threadIdx.x;
    int src = idx[k];
    unsigned short b = f2bf(x[(size_t)src * DIM + d]);
    cb[k * DIM + d] = b;
    float f = bf2f(b);
    red[d] = f * f;
    __syncthreads();
    for (int s = 128; s > 0; s >>= 1) {
        if (d < s) red[d] += red[d + s];
        __syncthreads();
    }
    if (d == 0) hn[k] = 0.5f * red[0];
}

// new c = sums/counts (NaN if empty, like reference); zero accumulators.
__global__ void k_finalize(float* __restrict__ sums, float* __restrict__ counts,
                           unsigned short* __restrict__ cb, float* __restrict__ hn) {
    __shared__ float red[256];
    int k = blockIdx.x, d = threadIdx.x;
    float cnt = counts[k];
    float mean = sums[k * DIM + d] / cnt;
    sums[k * DIM + d] = 0.0f;
    unsigned short b = f2bf(mean);
    cb[k * DIM + d] = b;
    float f = bf2f(b);
    red[d] = f * f;
    __syncthreads();
    for (int s = 128; s > 0; s >>= 1) {
        if (d < s) red[d] += red[d + s];
        __syncthreads();
    }
    if (d == 0) { hn[k] = 0.5f * red[0]; counts[k] = 0.0f; }
}

// ---------------------------------------------------------------------------
// Assignment + LDS-accumulated segment-sum.
// One block = 512 points (4 tiles of 128; 8 waves x 16 rows per tile).
// argmin ||x-c||^2 == argmax (x.c - 0.5||c||^2) -> WMMA GEMM + per-row argmax.
// LDS: c bf16 (64KB) + f32 cluster sums (128KB) + scratch (~17KB) ~= 210KB,
// legal only with CDNA5's 320KB-per-workgroup LDS.
#define TILES_PER_BLOCK 4
#define PTS_PER_BLOCK   (TILES_PER_BLOCK * 128)

__global__ void __launch_bounds__(256)
k_assign(const float* __restrict__ x, const unsigned short* __restrict__ cb,
         const float* __restrict__ hn, int* __restrict__ ids,
         float* __restrict__ gsums, float* __restrict__ gcounts) {
    extern __shared__ char smem[];
    unsigned short* ldsC    = (unsigned short*)smem;          //  65536 B: c bf16 [128][256]
    float*          ldsHN   = (float*)(smem + 65536);         //    512 B
    float*          ldsSums = (float*)(smem + 66048);         // 131072 B: f32 [128][256]
    float*          redV    = (float*)(smem + 197120);        //   8192 B: [128][16]
    int*            redI    = (int*)  (smem + 205312);        //   8192 B
    int*            ldsId   = (int*)  (smem + 213504);        //    512 B
    float*          ldsCnt  = (float*)(smem + 214016);        //    512 B
    // total 214528 B

    int tid = threadIdx.x;
    {   // stage centroids into LDS; zero LDS accumulators
        const uint4* src = (const uint4*)cb;
        uint4* dst = (uint4*)ldsC;
#pragma unroll
        for (int i = 0; i < 16; ++i) dst[tid + 256 * i] = src[tid + 256 * i];
        if (tid < KCL) { ldsHN[tid] = hn[tid]; ldsCnt[tid] = 0.0f; }
        for (int i = tid; i < KCL * DIM; i += 256) ldsSums[i] = 0.0f;
    }
    __syncthreads();

    int w = tid >> 5, lane = tid & 31;
    int h = lane >> 4, n = lane & 15;

    for (int t = 0; t < TILES_PER_BLOCK; ++t) {
        int tileBase = blockIdx.x * PTS_PER_BLOCK + t * 128;
        int rowTile0 = tileBase + w * 16;

        // Preload the 8 A-fragments (this wave's 16 x-rows, full D=256).
        Frag A[8];
        const float* xrow = x + (size_t)(rowTile0 + n) * DIM;
#pragma unroll
        for (int dc = 0; dc < 8; ++dc) load_a_frag(A[dc], xrow + dc * 32, h);

        float bv[8];
        int   bi[8];
#pragma unroll
        for (int v = 0; v < 8; ++v) { bv[v] = -3.0e38f; bi[v] = 0; }

        for (int ct = 0; ct < 8; ++ct) {       // 8 cluster tiles of 16
            v8f acc = {0.f, 0.f, 0.f, 0.f, 0.f, 0.f, 0.f, 0.f};
            const unsigned short* crow = ldsC + (ct * 16 + n) * DIM;
#pragma unroll
            for (int dc = 0; dc < 8; ++dc) {   // K = 256 in chunks of 32
                Frag B;                        // B 32x16: lane half h -> K h*16..h*16+15
                const uint4* bp = (const uint4*)(crow + dc * 32 + h * 16);
                B.q[0] = bp[0]; B.q[1] = bp[1];
                acc = __builtin_amdgcn_wmma_f32_16x16x32_bf16(
                    false, A[dc].v, false, B.v, (short)0, acc, false, false);
            }
            float hnc = ldsHN[ct * 16 + n];
#pragma unroll
            for (int v = 0; v < 8; ++v) {      // row M = v + h*8, cluster ct*16+n
                float sc = acc[v] - hnc;
                if (sc > bv[v]) { bv[v] = sc; bi[v] = ct * 16 + n; }
            }
        }
#pragma unroll
        for (int v = 0; v < 8; ++v) {
            int slot = w * 16 + v + h * 8;
            redV[slot * 16 + n] = bv[v];
            redI[slot * 16 + n] = bi[v];
        }
        __syncthreads();

        if (tid < 128) {                       // cross-lane argmax per row
            float best = -3.0e38f; int besti = 0;
#pragma unroll
            for (int c = 0; c < 16; ++c) {
                float v = redV[tid * 16 + c];
                int   i = redI[tid * 16 + c];
                if (v > best || (v == best && i < besti)) { best = v; besti = i; }
            }
            ldsId[tid] = besti;
            ids[tileBase + tid] = besti;
        }
        __syncthreads();

        {   // segment sums into LDS: thread tid owns dim tid (consecutive banks)
            const float* xb = x + (size_t)tileBase * DIM + tid;
            for (int p = 0; p < 128; ++p) {
                int id = ldsId[p];
                atomicAdd(&ldsSums[id * DIM + tid], xb[(size_t)p * DIM]);
            }
            if (tid < 128) atomicAdd(&ldsCnt[ldsId[tid]], 1.0f);
        }
        __syncthreads();                       // redV/redI/ldsId reused next tile
    }

    // flush per-block partials: 32768 global atomics per block (4x fewer total)
    for (int i = tid; i < KCL * DIM; i += 256)
        atomicAdd(&gsums[i], ldsSums[i]);
    if (tid < KCL) atomicAdd(&gcounts[tid], ldsCnt[tid]);
}

// ---------------------------------------------------------------------------
// logits = x @ c.T, online log-softmax, CE against stored ids.
// Per-block LDS tree reduction -> one atomic per block on d_out.
__global__ void __launch_bounds__(256)
k_loss(const float* __restrict__ x, const unsigned short* __restrict__ cb,
       const int* __restrict__ ids, float* __restrict__ out) {
    extern __shared__ char smem[];
    unsigned short* ldsC  = (unsigned short*)smem;        // 65536 B
    int*            ldsId = (int*)  (smem + 65536);       //   512 B
    float*          redM  = (float*)(smem + 66048);       //  8192 B
    float*          redS  = (float*)(smem + 74240);       //  8192 B
    float*          redT  = (float*)(smem + 82432);       //  8192 B
    float*          ldsLP = (float*)(smem + 90624);       //   512 B
    // total 91136 B

    int tid = threadIdx.x;
    {
        const uint4* src = (const uint4*)cb;
        uint4* dst = (uint4*)ldsC;
#pragma unroll
        for (int i = 0; i < 16; ++i) dst[tid + 256 * i] = src[tid + 256 * i];
        if (tid < 128) ldsId[tid] = ids[blockIdx.x * 128 + tid];
    }
    __syncthreads();

    int w = tid >> 5, lane = tid & 31;
    int h = lane >> 4, n = lane & 15;
    int rowTile0 = blockIdx.x * 128 + w * 16;

    Frag A[8];
    const float* xrow = x + (size_t)(rowTile0 + n) * DIM;
#pragma unroll
    for (int dc = 0; dc < 8; ++dc) load_a_frag(A[dc], xrow + dc * 32, h);

    int myid[8];
#pragma unroll
    for (int v = 0; v < 8; ++v) myid[v] = ldsId[w * 16 + v + h * 8];

    float m[8], s[8], t[8];
#pragma unroll
    for (int v = 0; v < 8; ++v) { m[v] = -3.0e38f; s[v] = 0.0f; t[v] = 0.0f; }

    for (int ct = 0; ct < 8; ++ct) {
        v8f acc = {0.f, 0.f, 0.f, 0.f, 0.f, 0.f, 0.f, 0.f};
        const unsigned short* crow = ldsC + (ct * 16 + n) * DIM;
#pragma unroll
        for (int dc = 0; dc < 8; ++dc) {
            Frag B;
            const uint4* bp = (const uint4*)(crow + dc * 32 + h * 16);
            B.q[0] = bp[0]; B.q[1] = bp[1];
            acc = __builtin_amdgcn_wmma_f32_16x16x32_bf16(
                false, A[dc].v, false, B.v, (short)0, acc, false, false);
        }
#pragma unroll
        for (int v = 0; v < 8; ++v) {          // online softmax per element slot
            float val = acc[v];
            if (ct * 16 + n == myid[v]) t[v] = val;
            float mn = fmaxf(m[v], val);
            s[v] = s[v] * __expf(m[v] - mn) + __expf(val - mn);
            m[v] = mn;
        }
    }
#pragma unroll
    for (int v = 0; v < 8; ++v) {
        int slot = w * 16 + v + h * 8;
        redM[slot * 16 + n] = m[v];
        redS[slot * 16 + n] = s[v];
        redT[slot * 16 + n] = t[v];
    }
    __syncthreads();

    if (tid < 128) {
        float M = -3.0e38f;
#pragma unroll
        for (int c = 0; c < 16; ++c) M = fmaxf(M, redM[tid * 16 + c]);
        float S = 0.0f, T = 0.0f;
#pragma unroll
        for (int c = 0; c < 16; ++c) {
            S += redS[tid * 16 + c] * __expf(redM[tid * 16 + c] - M);
            T += redT[tid * 16 + c];
        }
        ldsLP[tid] = -(T - M - __logf(S));
    }
    __syncthreads();
    for (int s2 = 64; s2 > 0; s2 >>= 1) {      // block tree reduction
        if (tid < s2) ldsLP[tid] += ldsLP[tid + s2];
        __syncthreads();
    }
    if (tid == 0) atomicAdd(out, ldsLP[0] * (1.0f / (float)NPT));
}

// ---------------------------------------------------------------------------
extern "C" void kernel_launch(void* const* d_in, const int* in_sizes, int n_in,
                              void* d_out, int out_size, void* d_ws, size_t ws_size,
                              hipStream_t stream) {
    const float* x        = (const float*)d_in[0];
    const int*   init_idx = (const int*)d_in[1];
    float*       out      = (float*)d_out;

    char* ws = (char*)d_ws;
    float*          sums   = (float*)(ws + 0);            // 131072 B
    float*          counts = (float*)(ws + 131072);       //    512 B
    float*          hn     = (float*)(ws + 131584);       //    512 B
    int*            ids    = (int*)  (ws + 132096);       // 131072 B
    unsigned short* cb     = (unsigned short*)(ws + 263168); // 65536 B

    dim3 blk(256);
    const size_t shAssign = 214528;  // c + hn + ldsSums + redV + redI + ids + cnt
    const size_t shLoss   = 91136;   // c + ids + redM + redS + redT + lp

    k_init<<<dim3(128), blk, 0, stream>>>(sums, counts, out);
    k_gather<<<dim3(KCL), blk, 0, stream>>>(x, init_idx, cb, hn);

    for (int it = 0; it < ITERS; ++it) {
        k_assign<<<dim3(NPT / PTS_PER_BLOCK), blk, shAssign, stream>>>(
            x, cb, hn, ids, sums, counts);
        k_finalize<<<dim3(KCL), blk, 0, stream>>>(sums, counts, cb, hn);
    }
    // ids now correspond to centroids BEFORE the last update; cb is the final c.
    k_loss<<<dim3(NPT / 128), blk, shLoss, stream>>>(x, cb, ids, out);
}
